// CrossAttentionFusion__45561013076033
// MI455X (gfx1250) — matrix operand order repaired and checked by reading, not compiled, not run on hardware
//
#include <hip/hip_runtime.h>

#define BB 4
#define CC 256
#define NN 4096

typedef __attribute__((ext_vector_type(16))) __bf16 v16bf;
typedef __attribute__((ext_vector_type(8)))  __bf16 v8bf;
typedef __attribute__((ext_vector_type(8)))  float  v8f;
typedef __attribute__((ext_vector_type(4)))  unsigned int v4u;
typedef __attribute__((ext_vector_type(8)))  int v8i;
typedef __attribute__((ext_vector_type(4)))  int v4i;

__device__ __forceinline__ v8f v8f_zero() {
  v8f z;
#pragma unroll
  for (int i = 0; i < 8; ++i) z[i] = 0.f;
  return z;
}

__device__ __forceinline__ v8bf ld8(const __bf16* p) {
  union { uint4 u; v8bf b; } c;
  c.u = *(const uint4*)p;
  return c.b;
}

// A-fragment (16x32) from row-major [m][k] storage: two contiguous 8-element
// runs per lane: k = 8*half..+8 and k = 16+8*half..+8.  `row` = &S[m][0].
__device__ __forceinline__ v16bf load_frag_A_rm(const __bf16* row, int half) {
  v8bf lo = ld8(row + 8 * half);
  v8bf hi = ld8(row + 16 + 8 * half);
  return __builtin_shufflevector(lo, hi, 0, 1, 2, 3, 4, 5, 6, 7,
                                 8, 9, 10, 11, 12, 13, 14, 15);
}

// B-fragment (32x16) from [n][k] storage: two contiguous 8-element runs per
// lane: k = 8*half..+8 and k = 16+8*half..+8.  `row` = &S[n][0].
__device__ __forceinline__ v16bf load_frag_B_rm(const __bf16* row, int half) {
  v8bf lo = ld8(row + 8 * half);
  v8bf hi = ld8(row + 16 + 8 * half);
  return __builtin_shufflevector(lo, hi, 0, 1, 2, 3, 4, 5, 6, 7,
                                 8, 9, 10, 11, 12, 13, 14, 15);
}

// Pack 8 f32 accumulator slots (+bias) into 8 contiguous bf16 and store b128.
__device__ __forceinline__ void store_row_bf16(__bf16* dst, v8f acc, float bias) {
  __bf16 tmp[8];
#pragma unroll
  for (int j = 0; j < 8; ++j) tmp[j] = (__bf16)(acc[j] + bias);
  uint4 u;
  __builtin_memcpy(&u, tmp, 16);
  *(uint4*)dst = u;
}

// Stage one 64-token x 32-channel A tile (channel-major source) into [m][k].
template <typename T>
__device__ __forceinline__ void stage_tile(__bf16 (*dst)[40], const T* srcB,
                                           int kc, int n0, int tid) {
#pragma unroll
  for (int it = 0; it < 8; ++it) {
    int i = tid + it * 256;
    int m = i & 63;
    int k = i >> 6;
    dst[m][k] = (__bf16)srcB[(size_t)(kc * 32 + k) * NN + (n0 + m)];
  }
}

// ---------------------------------------------------------------------------
// Tensor Data Mover: 2D tile load Global->LDS with row padding.
// D# per CDNA5 ISA.  data_size = 2 bytes.
// ---------------------------------------------------------------------------
__device__ __forceinline__ void tdm_load_2d(unsigned lds_off, const void* gptr,
                                            unsigned tile_d0, unsigned tile_d1,
                                            unsigned tensor_d0, unsigned tensor_d1,
                                            unsigned long long d0_stride,
                                            unsigned pad_interval_code,
                                            unsigned pad_amount_code) {
  unsigned long long ga = (unsigned long long)(uintptr_t)gptr;
  v4u g0;
  g0[0] = 1u;                                   // count=1, user descriptor
  g0[1] = lds_off;                              // lds_addr
  g0[2] = (unsigned)(ga & 0xffffffffu);         // global_addr[31:0]
  g0[3] = (unsigned)((ga >> 32) & 0x01ffffffu)  // global_addr[56:32]
        | (2u << 30);                           // type = 2 ("image")
  v8i g1;
  unsigned w0 = (1u << 16)                      // data_size = 1 -> 2 bytes
              | (1u << 20)                      // pad_enable
              | (pad_interval_code << 22)
              | (pad_amount_code << 25);
  g1[0] = (int)w0;
  g1[1] = (int)((tensor_d0 & 0xffffu) << 16);
  g1[2] = (int)(((tensor_d0 >> 16) & 0xffffu) | ((tensor_d1 & 0xffffu) << 16));
  g1[3] = (int)(((tensor_d1 >> 16) & 0xffffu) | ((tile_d0 & 0xffffu) << 16));
  g1[4] = (int)(tile_d1 & 0xffffu);             // tile_dim1; tile_dim2 = 0
  g1[5] = (int)(unsigned)(d0_stride & 0xffffffffull);
  g1[6] = (int)(unsigned)((d0_stride >> 32) & 0xffffull);
  g1[7] = 0;
  v4i z4 = {0, 0, 0, 0};
  v8i z8 = {0, 0, 0, 0, 0, 0, 0, 0};
  __builtin_amdgcn_tensor_load_to_lds(g0, g1, z4, z4, z8, 0);
}

// ---------------------------------------------------------------------------
// Kernel 0: fp32 -> bf16 weight conversion.
// ---------------------------------------------------------------------------
__global__ __launch_bounds__(256) void caf_cvt_w(const float* __restrict__ s,
                                                 __bf16* __restrict__ d, int n) {
  int i = (blockIdx.x * 256 + threadIdx.x) * 4;
  if (i < n) {
    float4 f = *(const float4*)(s + i);
    __bf16 t[4] = {(__bf16)f.x, (__bf16)f.y, (__bf16)f.z, (__bf16)f.w};
    uint2 u;
    __builtin_memcpy(&u, t, 8);
    *(uint2*)(d + i) = u;
  }
}

// ---------------------------------------------------------------------------
// Kernel 1: Q/K/V projection.  out[n,d] = sum_c feat[c,n] * W[d,c] + b[d]
// grid (N/64, 3, B), block 256 (8 waves).  Double-buffered A staging;
// 8 B-fragments hoisted per k-step.  V is stored transposed [C][N].
// ---------------------------------------------------------------------------
__global__ __launch_bounds__(256) void caf_qkv_proj(
    const float* __restrict__ trf, const float* __restrict__ cnn,
    const __bf16* __restrict__ Wqb, const float* __restrict__ bq,
    const __bf16* __restrict__ Wkb, const float* __restrict__ bk,
    const __bf16* __restrict__ Wvb, const float* __restrict__ bv,
    __bf16* __restrict__ Qo, __bf16* __restrict__ Ko, __bf16* __restrict__ Vt) {
  const int b     = blockIdx.z;
  const int which = blockIdx.y;  // 0:q(trf) 1:k(cnn) 2:v(cnn, transposed out)
  const int n0    = blockIdx.x * 64;

  const float*  src  = (which == 0) ? trf : cnn;
  const __bf16* W    = (which == 0) ? Wqb : (which == 1) ? Wkb : Wvb;
  const float*  bias = (which == 0) ? bq : (which == 1) ? bk : bv;

  const int tid  = threadIdx.x;
  const int wave = tid >> 5;
  const int lane = tid & 31;
  const int half = lane >> 4;
  const int mr   = lane & 15;
  const int rowTile = wave & 3;
  const int colBase = (wave >> 2) * 128;

  __shared__ __align__(16) __bf16 At[2][64][40];  // double-buffered [m][k]

  v8f acc[8];
#pragma unroll
  for (int t = 0; t < 8; ++t) acc[t] = v8f_zero();

  const float* srcB = src + (size_t)b * CC * NN;

  stage_tile(At[0], srcB, 0, n0, tid);

  for (int kc = 0; kc < 8; ++kc) {
    __syncthreads();
    if (kc + 1 < 8) stage_tile(At[(kc + 1) & 1], srcB, kc + 1, n0, tid);

    v16bf a = load_frag_A_rm(&At[kc & 1][16 * rowTile + mr][0], half);
    v16bf wfr[8];
#pragma unroll
    for (int t = 0; t < 8; ++t)
      wfr[t] = load_frag_B_rm(W + (size_t)(colBase + t * 16 + mr) * CC + kc * 32, half);
#pragma unroll
    for (int t = 0; t < 8; ++t)
      acc[t] = __builtin_amdgcn_wmma_f32_16x16x32_bf16(
          false, a, false, wfr[t], (short)0, acc[t], false, false);
  }

  if (which == 2) {
    // V stored transposed [C][N]: D-tile rows are contiguous in token index.
#pragma unroll
    for (int t = 0; t < 8; ++t) {
      int d = colBase + t * 16 + mr;
      __bf16* p = Vt + ((size_t)b * CC + d) * NN + n0 + 16 * rowTile + 8 * half;
      store_row_bf16(p, acc[t], bias[d]);
    }
  } else {
    __bf16* dst = (which == 0) ? Qo : Ko;
#pragma unroll
    for (int t = 0; t < 8; ++t) {
      int d = colBase + t * 16 + mr;
      float bsv = bias[d];
#pragma unroll
      for (int j = 0; j < 8; ++j) {
        int m = 16 * rowTile + j + 8 * half;
        dst[((size_t)b * NN + (n0 + m)) * CC + d] = (__bf16)(acc[t][j] + bsv);
      }
    }
  }
}

// ---------------------------------------------------------------------------
// Kernel 2: flash attention.  grid (N/64, B), block 128 (4 waves x 16 rows).
// Q fragments live in registers.  K tile and transposed V tile in separate
// LDS buffers, both staged by one pair of TDM descriptors per key block
// (2 block barriers per iteration).
// ---------------------------------------------------------------------------
#define KVST 264  // K tile row stride (256 + 8 pad)
#define VTST 40   // V^T tile row stride (32 + 8 pad)
#define PST  40

__global__ __launch_bounds__(128) void caf_flash_attn(
    const __bf16* __restrict__ Qg, const __bf16* __restrict__ Kg,
    const __bf16* __restrict__ Vt, __bf16* __restrict__ ATTt) {
  const int b  = blockIdx.y;
  const int n0 = blockIdx.x * 64;
  const int tid  = threadIdx.x;
  const int wave = tid >> 5;
  const int lane = tid & 31;
  const int half = lane >> 4;
  const int mr   = lane & 15;

  __shared__ __align__(16) __bf16 Ks[32][KVST];    // [key][dim]
  __shared__ __align__(16) __bf16 Vts[256][VTST];  // [dim][key]
  __shared__ __align__(16) __bf16 Ps[4][16][PST];

  const unsigned ks_off = (unsigned)(uintptr_t)&Ks[0][0];
  const unsigned vt_off = (unsigned)(uintptr_t)&Vts[0][0];

  // Hoist this wave's Q fragments: A-frag of row-major [N][C] is two b128s.
  v16bf qfr[8];
#pragma unroll
  for (int kc = 0; kc < 8; ++kc)
    qfr[kc] = load_frag_A_rm(
        Qg + ((size_t)b * NN + n0 + wave * 16 + mr) * CC + kc * 32, half);

  v8f O[16];
#pragma unroll
  for (int t = 0; t < 16; ++t) O[t] = v8f_zero();
  float mrun[8], lrun[8];
#pragma unroll
  for (int j = 0; j < 8; ++j) { mrun[j] = -1e30f; lrun[j] = 0.f; }

  const float scale = 0.0625f;  // 1/sqrt(256)

  for (int kb = 0; kb < NN / 32; ++kb) {
    __syncthreads();  // everyone done reading previous Ks/Vts
    if (wave == 0) {
      // K tile: 32 x 256 bf16, rows 512B = 128 DW -> interval code 6;
      // pad 16B = 4 DW -> amount code 3 (LDS stride 264).
      tdm_load_2d(ks_off, Kg + ((size_t)b * NN + kb * 32) * CC,
                  256, 32, 256, 32, 256ull, 6, 3);
      // V^T tile: 256 x 32 bf16, rows 64B = 16 DW -> interval code 3;
      // pad 4 DW -> amount code 3 (LDS stride 40).
      tdm_load_2d(vt_off, Vt + (size_t)b * CC * NN + kb * 32,
                  32, 256, 32, 256, (unsigned long long)NN, 3, 3);
      __builtin_amdgcn_s_wait_tensorcnt(0);
      if (kb + 1 < NN / 32) {
        __builtin_prefetch(Kg + ((size_t)b * NN + (kb + 1) * 32) * CC, 0, 0);
        __builtin_prefetch(Vt + (size_t)b * CC * NN + (kb + 1) * 32, 0, 0);
      }
    }
    __syncthreads();

    // S = Q * K^T : B-frag of [key][dim] rows is two 16B (b128) runs.
    v8f S0 = v8f_zero(), S1 = v8f_zero();
#pragma unroll
    for (int kc = 0; kc < 8; ++kc) {
      v16bf b0 = load_frag_B_rm(&Ks[mr][kc * 32], half);
      v16bf b1 = load_frag_B_rm(&Ks[16 + mr][kc * 32], half);
      S0 = __builtin_amdgcn_wmma_f32_16x16x32_bf16(false, qfr[kc], false, b0,
                                                   (short)0, S0, false, false);
      S1 = __builtin_amdgcn_wmma_f32_16x16x32_bf16(false, qfr[kc], false, b1,
                                                   (short)0, S1, false, false);
    }

    // Online softmax (row j + 8*half lives within one 16-lane half).
    float alpha[8];
#pragma unroll
    for (int j = 0; j < 8; ++j) {
      float s0 = S0[j] * scale, s1 = S1[j] * scale;
      float cm = fmaxf(s0, s1);
#pragma unroll
      for (int off = 8; off >= 1; off >>= 1)
        cm = fmaxf(cm, __shfl_xor(cm, off, 32));
      float mnew = fmaxf(mrun[j], cm);
      float p0 = __expf(s0 - mnew), p1 = __expf(s1 - mnew);
      float rs = p0 + p1;
#pragma unroll
      for (int off = 8; off >= 1; off >>= 1) rs += __shfl_xor(rs, off, 32);
      alpha[j] = __expf(mrun[j] - mnew);
      lrun[j] = lrun[j] * alpha[j] + rs;
      mrun[j] = mnew;
      Ps[wave][j + 8 * half][mr]      = (__bf16)p0;
      Ps[wave][j + 8 * half][16 + mr] = (__bf16)p1;
    }
    // Same-wave LDS RAW: DS is in-order per wave; just stop compiler reorder.
    __builtin_amdgcn_wave_barrier();

#pragma unroll
    for (int t = 0; t < 16; ++t)
#pragma unroll
      for (int j = 0; j < 8; ++j) O[t][j] *= alpha[j];

    // O += P * V: P A-frags from row-major Ps; V B-frags from [dim][key] rows.
    v16bf pa = load_frag_A_rm(&Ps[wave][mr][0], half);
#pragma unroll
    for (int t = 0; t < 16; ++t) {
      v16bf vb = load_frag_B_rm(&Vts[t * 16 + mr][0], half);
      O[t] = __builtin_amdgcn_wmma_f32_16x16x32_bf16(false, pa, false, vb,
                                                     (short)0, O[t], false, false);
    }
  }

  // Attended output stored transposed [C][N]: contiguous b128 stores.
#pragma unroll
  for (int t = 0; t < 16; ++t) {
    __bf16 tmp[8];
#pragma unroll
    for (int j = 0; j < 8; ++j) tmp[j] = (__bf16)(O[t][j] / lrun[j]);
    uint4 u;
    __builtin_memcpy(&u, tmp, 16);
    __bf16* p = ATTt + ((size_t)b * CC + t * 16 + mr) * NN + n0 + wave * 16 + 8 * half;
    *(uint4*)p = u;
  }
}

// ---------------------------------------------------------------------------
// Kernel 3: fused 1x1 conv.  out[b,d,n] = sum_{c<512} X[c,n]*Wf[d,c] + bf[d]
// X = [trf (fp32, [C,N]); attended (bf16, [C,N])].  grid (N/64, B), block 256.
// ---------------------------------------------------------------------------
__global__ __launch_bounds__(256) void caf_fuse_conv(
    const float* __restrict__ trf, const __bf16* __restrict__ ATTt,
    const __bf16* __restrict__ Wfb, const float* __restrict__ bfv,
    float* __restrict__ out) {
  const int b  = blockIdx.y;
  const int n0 = blockIdx.x * 64;
  const int tid  = threadIdx.x;
  const int wave = tid >> 5;
  const int lane = tid & 31;
  const int half = lane >> 4;
  const int mr   = lane & 15;
  const int rowTile = wave & 3;
  const int colBase = (wave >> 2) * 128;

  __shared__ __align__(16) __bf16 At[2][64][40];

  v8f acc[8];
#pragma unroll
  for (int t = 0; t < 8; ++t) acc[t] = v8f_zero();

  const float*  tsrc = trf + (size_t)b * CC * NN;
  const __bf16* asrc = ATTt + (size_t)b * CC * NN;

  stage_tile(At[0], tsrc, 0, n0, tid);

  for (int kc = 0; kc < 16; ++kc) {
    __syncthreads();
    int nk = kc + 1;
    if (nk < 16) {
      if (nk < 8) stage_tile(At[nk & 1], tsrc, nk, n0, tid);
      else        stage_tile(At[nk & 1], asrc, nk - 8, n0, tid);
    }

    v16bf a = load_frag_A_rm(&At[kc & 1][16 * rowTile + mr][0], half);
    v16bf wfr[8];
#pragma unroll
    for (int t = 0; t < 8; ++t)
      wfr[t] = load_frag_B_rm(
          Wfb + (size_t)(colBase + t * 16 + mr) * (2 * CC) + kc * 32, half);
#pragma unroll
    for (int t = 0; t < 8; ++t)
      acc[t] = __builtin_amdgcn_wmma_f32_16x16x32_bf16(
          false, a, false, wfr[t], (short)0, acc[t], false, false);
  }

  // out is [C][N] fp32 -> D-tile rows contiguous in token index.
#pragma unroll
  for (int t = 0; t < 8; ++t) {
    int d = colBase + t * 16 + mr;
    float bsv = bfv[d];
    float4 o0 = {acc[t][0] + bsv, acc[t][1] + bsv, acc[t][2] + bsv, acc[t][3] + bsv};
    float4 o1 = {acc[t][4] + bsv, acc[t][5] + bsv, acc[t][6] + bsv, acc[t][7] + bsv};
    float* p = out + ((size_t)b * CC + d) * NN + n0 + 16 * rowTile + 8 * half;
    *(float4*)p = o0;
    *(float4*)(p + 4) = o1;
  }
}

// ---------------------------------------------------------------------------
extern "C" void kernel_launch(void* const* d_in, const int* in_sizes, int n_in,
                              void* d_out, int out_size, void* d_ws, size_t ws_size,
                              hipStream_t stream) {
  const float* trf = (const float*)d_in[0];
  const float* cnn = (const float*)d_in[1];
  const float* Wq  = (const float*)d_in[2];
  const float* bq  = (const float*)d_in[3];
  const float* Wk  = (const float*)d_in[4];
  const float* bk  = (const float*)d_in[5];
  const float* Wv  = (const float*)d_in[6];
  const float* bv  = (const float*)d_in[7];
  const float* Wf  = (const float*)d_in[8];
  const float* bf  = (const float*)d_in[9];
  float* out = (float*)d_out;

  const size_t E = (size_t)BB * NN * CC;
  __bf16* Q    = (__bf16*)d_ws;  // [B][N][C]
  __bf16* K    = Q + E;          // [B][N][C]
  __bf16* Vt   = K + E;          // [B][C][N]
  __bf16* ATTt = Vt + E;         // [B][C][N]
  __bf16* Wqb  = ATTt + E;
  __bf16* Wkb  = Wqb + CC * CC;
  __bf16* Wvb  = Wkb + CC * CC;
  __bf16* Wfb  = Wvb + CC * CC;  // [C][2C]

  caf_cvt_w<<<CC * CC / 1024, 256, 0, stream>>>(Wq, Wqb, CC * CC);
  caf_cvt_w<<<CC * CC / 1024, 256, 0, stream>>>(Wk, Wkb, CC * CC);
  caf_cvt_w<<<CC * CC / 1024, 256, 0, stream>>>(Wv, Wvb, CC * CC);
  caf_cvt_w<<<CC * 2 * CC / 1024, 256, 0, stream>>>(Wf, Wfb, CC * 2 * CC);

  caf_qkv_proj<<<dim3(NN / 64, 3, BB), 256, 0, stream>>>(
      trf, cnn, Wqb, bq, Wkb, bk, Wvb, bv, Q, K, Vt);
  caf_flash_attn<<<dim3(NN / 64, BB), 128, 0, stream>>>(Q, K, Vt, ATTt);
  caf_fuse_conv<<<dim3(NN / 64, BB), 256, 0, stream>>>(trf, ATTt, Wfb, bf, out);
}
